// PointNet_Plus_Module_54494545052191
// MI455X (gfx1250) — compile-verified, matrix operand order checked by reading, not run on hardware
//
#include <hip/hip_runtime.h>
#include <hip/hip_bf16.h>

typedef __attribute__((ext_vector_type(16))) _Float16 v16h;
typedef __attribute__((ext_vector_type(8)))  float    v8f;

#define BATCH   8
#define NPTS    2048
#define NSAMP   64
#define R2      0.36f
#define K2WAVES 8

union FragH { float4 f4[2]; v16h h; };

// Wave-local LDS fence: CDNA5 LDS ops from one wave execute in order, so a
// dscnt-drain plus a compiler memory fence is sufficient to order producer
// stores against consumer loads within the same wave (no block barrier).
__device__ __forceinline__ void wave_lds_fence() {
    asm volatile("s_wait_dscnt 0x0" ::: "memory");
}

// ---------------------------------------------------------------------------
// Kernel 1: ball query. One wave32 per query point. Collect the first 64
// in-radius indices in ascending order via ballot + prefix popcount.
// ---------------------------------------------------------------------------
__global__ __launch_bounds__(256) void ball_query_kernel(
    const float* __restrict__ x, int* __restrict__ idxbuf)
{
    const int wave  = blockIdx.x * (blockDim.x >> 5) + (threadIdx.x >> 5);
    const int lane  = threadIdx.x & 31;
    const int b     = wave >> 11;          // / 2048
    const int n     = wave & (NPTS - 1);
    const float* xb = x + (size_t)b * NPTS * 3;

    const float qx = xb[n * 3 + 0];
    const float qy = xb[n * 3 + 1];
    const float qz = xb[n * 3 + 2];

    int* __restrict__ out = idxbuf + (size_t)wave * NSAMP;

    int count = 0;
    int first = 0;
    bool haveFirst = false;

    for (int chunk = 0; chunk < NPTS / 32; ++chunk) {
        const int m = chunk * 32 + lane;
        const float dx = xb[m * 3 + 0] - qx;
        const float dy = xb[m * 3 + 1] - qy;
        const float dz = xb[m * 3 + 2] - qz;
        const float d2 = fmaf(dx, dx, fmaf(dy, dy, dz * dz));
        const bool  in = (d2 <= R2);

        const unsigned mask = (unsigned)__ballot(in);   // wave32: bits [31:0]
        if (!haveFirst && mask) {
            first = chunk * 32 + __builtin_ctz(mask);
            haveFirst = true;
        }
        const int prefix = __popc(mask & ((1u << lane) - 1u));
        const int pos    = count + prefix;
        if (in && pos < NSAMP) out[pos] = m;
        count += __popc(mask);                          // uniform across wave
        if (count >= NSAMP) break;
    }
    // Pad remainder with the first in-radius index (self always qualifies).
    for (int pos = count + lane; pos < NSAMP; pos += 32) out[pos] = first;
}

// ---------------------------------------------------------------------------
// Kernel 2: gather + MLP(3->64->128, ReLU) + max over samples, WMMA layer 2.
// One query per wave; 8 waves / block; single block barrier after staging.
// ---------------------------------------------------------------------------
__global__ __launch_bounds__(256) void pointnet_mlp_kernel(
    const float* __restrict__ x,
    const float* __restrict__ W1, const float* __restrict__ b1,
    const float* __restrict__ W2, const float* __restrict__ b2,
    const int* __restrict__ idxbuf, float* __restrict__ outp)
{
    __shared__ __align__(16) float    sW1[64 * 3];
    __shared__ __align__(16) float    sb1[64];
    __shared__ __align__(16) _Float16 sW2[128 * 64];            // [n][k] row-major
    __shared__ __align__(16) float    sG [K2WAVES * NSAMP * 3]; // gathered coords
    __shared__ __align__(16) _Float16 sH1[K2WAVES * 16 * 64];   // per-wave h1 M-tile

    const int tid    = threadIdx.x;
    const int waveId = tid >> 5;
    const int lane   = tid & 31;
    const int query  = blockIdx.x * K2WAVES + waveId;
    const int b      = query >> 11;
    const int n      = query & (NPTS - 1);
    const float* xb  = x + (size_t)b * NPTS * 3;
    const int* idx   = idxbuf + (size_t)query * NSAMP;

    // Cooperative weight staging (W2 converted to f16 for WMMA).
    for (int i = tid; i < 64 * 3;   i += 256) sW1[i] = W1[i];
    for (int i = tid; i < 64;       i += 256) sb1[i] = b1[i];
    for (int i = tid; i < 128 * 64; i += 256) sW2[i] = (_Float16)W2[i];

    // Gather the 64 neighbor coordinates into this wave's private LDS slab.
    float* g = sG + waveId * (NSAMP * 3);
    for (int s = lane; s < NSAMP; s += 32) {
        const int i = idx[s];
        g[s * 3 + 0] = xb[i * 3 + 0];
        g[s * 3 + 1] = xb[i * 3 + 1];
        g[s * 3 + 2] = xb[i * 3 + 2];
    }

    __syncthreads();   // single block barrier: weights + gather visible

    const int row = lane & 15;          // M (A) / N (B) index within tile
    const int kb  = (lane >> 4) * 8;    // lanes 0-15 -> K 0-7/16-23, 16-31 -> K 8-15/24-31

    float bias2[8];
    #pragma unroll
    for (int nt = 0; nt < 8; ++nt) bias2[nt] = b2[nt * 16 + row];

    float runmax[8];
    #pragma unroll
    for (int nt = 0; nt < 8; ++nt) runmax[nt] = 0.0f;   // ReLU output >= 0

    _Float16* hslab = sH1 + waveId * (16 * 64);

    for (int mt = 0; mt < 4; ++mt) {
        // ---- Layer 1 (K=3) on VALU: 16 samples x 64 features, f16 to LDS ----
        const int   s  = mt * 16 + row;
        const int   fh = (lane >> 4) * 32;              // feature half
        const float px = g[s * 3 + 0];
        const float py = g[s * 3 + 1];
        const float pz = g[s * 3 + 2];
        _Float16* hrow = hslab + row * 64 + fh;
        #pragma unroll
        for (int j = 0; j < 32; ++j) {
            const int f = fh + j;
            float v = fmaf(px, sW1[f * 3 + 0],
                      fmaf(py, sW1[f * 3 + 1],
                      fmaf(pz, sW1[f * 3 + 2], sb1[f])));
            hrow[j] = (_Float16)fmaxf(v, 0.0f);
        }
        wave_lds_fence();   // wave-local: h1 stores ordered before fragment loads

        // ---- A fragments for this M-tile (K=0..31 and K=32..63) ----
        FragH a0, a1;
        const _Float16* ap = hslab + row * 64;
        a0.f4[0] = *(const float4*)(ap + kb);
        a0.f4[1] = *(const float4*)(ap + kb + 16);
        a1.f4[0] = *(const float4*)(ap + 32 + kb);
        a1.f4[1] = *(const float4*)(ap + 32 + kb + 16);

        // ---- Layer 2: 8 N-tiles x 2 K-steps of v_wmma_f32_16x16x32_f16 ----
        #pragma unroll
        for (int nt = 0; nt < 8; ++nt) {
            const _Float16* wp = sW2 + (nt * 16 + row) * 64;
            FragH fb0, fb1;
            fb0.f4[0] = *(const float4*)(wp + kb);
            fb0.f4[1] = *(const float4*)(wp + kb + 16);
            fb1.f4[0] = *(const float4*)(wp + 32 + kb);
            fb1.f4[1] = *(const float4*)(wp + 32 + kb + 16);

            v8f c = {};
            c = __builtin_amdgcn_wmma_f32_16x16x32_f16(
                    false, a0.h, false, fb0.h, (short)0, c, false, false);
            c = __builtin_amdgcn_wmma_f32_16x16x32_f16(
                    false, a1.h, false, fb1.h, (short)0, c, false, false);

            // bias + ReLU + max over the 8 sample-rows held in this lane
            float tmax = 0.0f;
            #pragma unroll
            for (int rr = 0; rr < 8; ++rr) tmax = fmaxf(tmax, c[rr] + bias2[nt]);
            runmax[nt] = fmaxf(runmax[nt], tmax);
        }
        // WAR on hslab for next iteration is safe: LDS ops of a wave execute
        // in order; keep a compiler fence so stores aren't hoisted above loads.
        asm volatile("" ::: "memory");
    }

    // Merge the two M-halves (lane L holds rows 0-7, lane L+16 rows 8-15).
    #pragma unroll
    for (int nt = 0; nt < 8; ++nt) {
        float v = runmax[nt];
        v = fmaxf(v, __shfl_xor(v, 16, 32));
        if (lane < 16) {
            const int c = nt * 16 + lane;
            outp[((size_t)b * 128 + c) * NPTS + n] = v;
        }
    }
}

// ---------------------------------------------------------------------------
extern "C" void kernel_launch(void* const* d_in, const int* in_sizes, int n_in,
                              void* d_out, int out_size, void* d_ws, size_t ws_size,
                              hipStream_t stream)
{
    const float* x  = (const float*)d_in[0];
    const float* W1 = (const float*)d_in[1];
    const float* b1 = (const float*)d_in[2];
    const float* W2 = (const float*)d_in[3];
    const float* b2 = (const float*)d_in[4];
    float* out = (float*)d_out;
    int* idxbuf = (int*)d_ws;                 // 16384 * 64 ints = 4 MB

    const int nQueries = BATCH * NPTS;        // 16384
    const int nBlocks  = nQueries / (256 / 32);     // 2048

    ball_query_kernel<<<nBlocks, 256, 0, stream>>>(x, idxbuf);
    pointnet_mlp_kernel<<<nQueries / K2WAVES, 256, 0, stream>>>(
        x, W1, b1, W2, b2, idxbuf, out);
}
